// Clusterator_23562190585856
// MI455X (gfx1250) — matrix operand order repaired and checked by reading, not compiled, not run on hardware
//
#include <hip/hip_runtime.h>

#define N_ROWS   100000
#define N_PAD    100096          /* 782 * 128 */
#define DIM      512
#define KCL      256
#define NUM_ITERS 11             /* 10 detached + 1 output iteration */
#define CHUNK    4352            /* N_PAD / 23, multiple of 32 */

#define BROW   40                /* padded LDS row stride (elements): 80B, conflict-free */
#define BSTAGE (KCL * BROW)      /* 10240 elements per staging buffer */
#define GDW    128               /* d-columns per mu-gemm block */

typedef __attribute__((ext_vector_type(16))) __bf16       bf16x16;
typedef __attribute__((ext_vector_type(8)))  float        f32x8;
typedef __attribute__((ext_vector_type(4)))  unsigned int u32x4;

union Frag { bf16x16 v; u32x4 u[2]; };

__device__ __forceinline__ unsigned short f2bf(float f) {
  union { float f; unsigned int u; } x; x.f = f;
  unsigned int u = x.u;
  u += 0x7fffu + ((u >> 16) & 1u);        /* round-to-nearest-even */
  return (unsigned short)(u >> 16);
}

__device__ __forceinline__ f32x8 zero8() {
  f32x8 v;
#pragma unroll
  for (int i = 0; i < 8; ++i) v[i] = 0.f;
  return v;
}

/* ---- normalize rows of a [rows, 512] f32 matrix, emit bf16; pad rows -> 0 ---- */
__global__ __launch_bounds__(256)
void row_norm_kernel(const float* __restrict__ in, unsigned short* __restrict__ out, int valid)
{
  const int row = blockIdx.x;
  const int t   = threadIdx.x;
  if (row >= valid) {                      /* uniform per block */
    out[(size_t)row * DIM + t]       = 0;
    out[(size_t)row * DIM + 256 + t] = 0;
    return;
  }
  const float a = in[(size_t)row * DIM + t];
  const float b = in[(size_t)row * DIM + 256 + t];
  float ss = a * a + b * b;
#pragma unroll
  for (int mask = 1; mask < 32; mask <<= 1) ss += __shfl_xor(ss, mask, 32);
  __shared__ float red[8];
  if ((t & 31) == 0) red[t >> 5] = ss;
  __syncthreads();
  float tot = 0.f;
#pragma unroll
  for (int i = 0; i < 8; ++i) tot += red[i];
  const float inv = 1.0f / (sqrtf(tot) + 1e-6f);
  out[(size_t)row * DIM + t]       = f2bf(a * inv);
  out[(size_t)row * DIM + 256 + t] = f2bf(b * inv);
}

/* ---- bf16 [N_PAD, DIM] -> [DIM, N_PAD] transpose ---- */
__global__ __launch_bounds__(256)
void transpose_kernel(const unsigned short* __restrict__ in, unsigned short* __restrict__ out)
{
  __shared__ unsigned short tile[32][33];
  const int d  = blockIdx.x * 32 + threadIdx.x;
  const int n0 = blockIdx.y * 32;
  for (int j = threadIdx.y; j < 32; j += 8)
    tile[j][threadIdx.x] = in[(size_t)(n0 + j) * DIM + d];
  __syncthreads();
  const int n   = n0 + threadIdx.x;
  const int dd0 = blockIdx.x * 32;
  for (int j = threadIdx.y; j < 32; j += 8)
    out[(size_t)(dd0 + j) * N_PAD + n] = tile[threadIdx.x][j];
}

__global__ void zero_f32(float* __restrict__ p, int n)
{
  const int i = blockIdx.x * 256 + threadIdx.x;
  if (i < n) p[i] = 0.f;
}

/* ---- phase 1: dist = data_n @ mu_n^T, softmax rows, emit r_t (bf16, transposed),
        colsum atomics, optional f32 r output.  One wave = 16 data rows x all 256 k.
        mu_n d-slices are staged in LDS (double-buffered, shared by all 8 waves). ---- */
__global__ __launch_bounds__(256)
void dist_softmax_kernel(const unsigned short* __restrict__ data_n,
                         const unsigned short* __restrict__ mu_n,
                         unsigned short* __restrict__ r_t,
                         float* __restrict__ colsum,
                         const float* __restrict__ temp_ptr,
                         float* __restrict__ r_out, int write_r)
{
  /* 64 KB: first used as 2 x 20KB B-staging buffers, then as 8 x 8KB transpose space */
  __shared__ __align__(16) unsigned short smem[32768];
  const int wave = threadIdx.x >> 5;
  const int lane = threadIdx.x & 31;
  const int m    = lane & 15;
  const int h    = lane >> 4;
  const int n0   = blockIdx.x * 128 + wave * 16;
  const float temp = temp_ptr[0];

  f32x8 acc[16];
#pragma unroll
  for (int kt = 0; kt < 16; ++kt) acc[kt] = zero8();

  const unsigned short* arow = data_n + (size_t)(n0 + m) * DIM;

  auto stage = [&](int p, int dt) {        /* all 256 threads: one mu_n row-slice each */
    const unsigned short* src = mu_n + (size_t)threadIdx.x * DIM + dt * 32;
    unsigned short* dst = smem + p * BSTAGE + threadIdx.x * BROW;
    *(u32x4*)(dst)      = *(const u32x4*)(src);
    *(u32x4*)(dst + 8)  = *(const u32x4*)(src + 8);
    *(u32x4*)(dst + 16) = *(const u32x4*)(src + 16);
    *(u32x4*)(dst + 24) = *(const u32x4*)(src + 24);
  };
  auto loadA = [&](Frag& a, int dt) {
    a.u[0] = *(const u32x4*)(arow + dt * 32 + h * 8);
    a.u[1] = *(const u32x4*)(arow + dt * 32 + 16 + h * 8);
  };
  auto loadB = [&](Frag& b, const unsigned short* bs, int kt) {
    const unsigned short* br = bs + (kt * 16 + m) * BROW + h * 8;
    b.u[0] = *(const u32x4*)(br);
    b.u[1] = *(const u32x4*)(br + 16);
  };

  Frag aa[2];
  stage(0, 0);
  loadA(aa[0], 0);
  __syncthreads();

  for (int dt = 0; dt < 16; ++dt) {
    const int p = dt & 1;
    if (dt < 15) { stage(p ^ 1, dt + 1); loadA(aa[p ^ 1], dt + 1); }
    const unsigned short* bs = smem + p * BSTAGE;
    /* depth-2 B-fragment pipeline: loads for kt+1 issued before the WMMA for kt */
    Frag bf[2];
    loadB(bf[0], bs, 0);
#pragma unroll
    for (int kt = 0; kt < 16; ++kt) {
      if (kt < 15) loadB(bf[(kt + 1) & 1], bs, kt + 1);
      acc[kt] = __builtin_amdgcn_wmma_f32_16x16x32_bf16(
          false, aa[p].v, false, bf[kt & 1].v, (short)0, acc[kt], false, false);
    }
    __syncthreads();                       /* staging of p^1 done, reads of p done */
  }

  /* dist is a cosine similarity (<=1), so exp(temp*dist) <= e^30: no max-shift needed */
  float rowsum[8];
#pragma unroll
  for (int i = 0; i < 8; ++i) rowsum[i] = 0.f;
#pragma unroll
  for (int kt = 0; kt < 16; ++kt) {
#pragma unroll
    for (int i = 0; i < 8; ++i) {
      float e = __expf(temp * acc[kt][i]);
      if (n0 + h * 8 + i >= N_ROWS) e = 0.f;   /* mask pad rows */
      acc[kt][i] = e;
      rowsum[i] += e;
    }
  }
#pragma unroll
  for (int mask = 1; mask < 16; mask <<= 1) {
#pragma unroll
    for (int i = 0; i < 8; ++i) rowsum[i] += __shfl_xor(rowsum[i], mask, 32);
  }
  float rinv[8];
#pragma unroll
  for (int i = 0; i < 8; ++i) rinv[i] = 1.0f / (rowsum[i] > 0.f ? rowsum[i] : 1.0f);

  unsigned short* smw = smem + wave * 4096;    /* private 8KB transpose region */
#pragma unroll
  for (int kt = 0; kt < 16; ++kt) {
    const int k = kt * 16 + m;
    float cp = 0.f;
#pragma unroll
    for (int i = 0; i < 8; ++i) {
      const float rv = acc[kt][i] * rinv[i];
      const int   nl = h * 8 + i;
      smw[k * 16 + nl] = f2bf(rv);
      cp += rv;
      if (write_r) {
        const int row = n0 + nl;
        if (row < N_ROWS) r_out[(size_t)row * KCL + k] = rv;
      }
    }
    cp += __shfl_xor(cp, 16, 32);            /* combine both 8-row halves */
    if (h == 0) atomicAdd(&colsum[k], cp);
  }
  __syncthreads();

  /* coalesced transposed store: each lane writes 8 k-rows of 16 contiguous n */
#pragma unroll
  for (int j = 0; j < 8; ++j) {
    const int k = j * 32 + lane;
    const u32x4 lo = *(const u32x4*)(smw + k * 16);
    const u32x4 hi = *(const u32x4*)(smw + k * 16 + 8);
    unsigned short* dst = r_t + (size_t)k * N_PAD + n0;
    *(u32x4*)(dst)     = lo;
    *(u32x4*)(dst + 8) = hi;
  }
}

/* ---- phase 2: mu_acc[k,d] += sum_n r_t[k,n] * data_t[d,n].
        data_t tiles staged in LDS (shared by all waves), double-buffered;
        N split over blockIdx.z chunks, partials merged via f32 atomics. ---- */
__global__ __launch_bounds__(256)
void mu_gemm_kernel(const unsigned short* __restrict__ r_t,
                    const unsigned short* __restrict__ data_t,
                    float* __restrict__ mu_acc)
{
  __shared__ __align__(16) unsigned short bsm[2][GDW * BROW];   /* 2 x 10KB */
  const int wave = threadIdx.x >> 5;
  const int lane = threadIdx.x & 31;
  const int m    = lane & 15;
  const int h    = lane >> 4;
  const int k0   = blockIdx.x * 128 + wave * 16;
  const int d0   = blockIdx.y * GDW;
  const int nbeg = blockIdx.z * CHUNK;

  f32x8 acc[8];
#pragma unroll
  for (int dt = 0; dt < 8; ++dt) acc[dt] = zero8();

  const unsigned short* arow = r_t + (size_t)(k0 + m) * N_PAD;

  auto stage = [&](int p, int nb) {        /* 256 threads: 128 rows x 32 n, 32B each */
    const int row  = threadIdx.x >> 1;
    const int half = threadIdx.x & 1;
    const unsigned short* src = data_t + (size_t)(d0 + row) * N_PAD + nb + half * 16;
    unsigned short* dst = bsm[p] + row * BROW + half * 16;
    *(u32x4*)(dst)     = *(const u32x4*)(src);
    *(u32x4*)(dst + 8) = *(const u32x4*)(src + 8);
  };
  auto loadA = [&](Frag& a, int nb) {
    a.u[0] = *(const u32x4*)(arow + nb + h * 8);
    a.u[1] = *(const u32x4*)(arow + nb + 16 + h * 8);
  };
  auto loadB = [&](Frag& b, const unsigned short* bs, int dt) {
    const unsigned short* br = bs + (dt * 16 + m) * BROW + h * 8;
    b.u[0] = *(const u32x4*)(br);
    b.u[1] = *(const u32x4*)(br + 16);
  };

  Frag aa[2];
  stage(0, nbeg);
  loadA(aa[0], nbeg);
  __syncthreads();

  const int NT = CHUNK / 32;               /* 136 tiles */
  for (int t = 0; t < NT; ++t) {
    const int p = t & 1;
    if (t + 1 < NT) { stage(p ^ 1, nbeg + (t + 1) * 32); loadA(aa[p ^ 1], nbeg + (t + 1) * 32); }
    const unsigned short* bs = bsm[p];
    /* depth-2 B-fragment pipeline */
    Frag bf[2];
    loadB(bf[0], bs, 0);
#pragma unroll
    for (int dt = 0; dt < 8; ++dt) {
      if (dt < 7) loadB(bf[(dt + 1) & 1], bs, dt + 1);
      acc[dt] = __builtin_amdgcn_wmma_f32_16x16x32_bf16(
          false, aa[p].v, false, bf[dt & 1].v, (short)0, acc[dt], false, false);
    }
    __syncthreads();
  }
#pragma unroll
  for (int dt = 0; dt < 8; ++dt) {
#pragma unroll
    for (int i = 0; i < 8; ++i)
      atomicAdd(&mu_acc[(size_t)(k0 + h * 8 + i) * DIM + d0 + dt * 16 + m], acc[dt][i]);
  }
}

/* ---- phase 3: mu = mu_acc / colsum[k]; optionally also to d_out ---- */
__global__ __launch_bounds__(256)
void finalize_kernel(const float* __restrict__ mu_acc, const float* __restrict__ colsum,
                     float* __restrict__ mu, float* __restrict__ out_mu)
{
  const int idx = blockIdx.x * 256 + threadIdx.x;     /* grid covers exactly K*D */
  const float v = mu_acc[idx] / colsum[idx >> 9];
  mu[idx] = v;
  if (out_mu) out_mu[idx] = v;
}

extern "C" void kernel_launch(void* const* d_in, const int* in_sizes, int n_in,
                              void* d_out, int out_size, void* d_ws, size_t ws_size,
                              hipStream_t stream)
{
  (void)in_sizes; (void)n_in; (void)out_size; (void)ws_size;
  const float* embeds = (const float*)d_in[0];
  const float* ctemp  = (const float*)d_in[1];
  const float* init   = (const float*)d_in[2];
  float* out_mu = (float*)d_out;                 /* [K, D] */
  float* out_r  = (float*)d_out + KCL * DIM;     /* [N, K] */

  char* ws = (char*)d_ws;
  size_t off = 0;
  auto take = [&](size_t bytes) { char* p = ws + off; off += (bytes + 255) & ~(size_t)255; return p; };
  unsigned short* data_n = (unsigned short*)take((size_t)N_PAD * DIM * 2);
  unsigned short* data_t = (unsigned short*)take((size_t)DIM * N_PAD * 2);
  unsigned short* r_t    = (unsigned short*)take((size_t)KCL * N_PAD * 2);
  unsigned short* mu_n   = (unsigned short*)take((size_t)KCL * DIM * 2);
  float* mu     = (float*)take((size_t)KCL * DIM * 4);
  float* mu_acc = (float*)take((size_t)KCL * DIM * 4);
  float* colsum = (float*)take((size_t)KCL * 4);

  hipMemcpyAsync(mu, init, (size_t)KCL * DIM * sizeof(float),
                 hipMemcpyDeviceToDevice, stream);
  row_norm_kernel<<<N_PAD, 256, 0, stream>>>(embeds, data_n, N_ROWS);
  transpose_kernel<<<dim3(DIM / 32, N_PAD / 32), dim3(32, 8), 0, stream>>>(data_n, data_t);

  for (int it = 0; it < NUM_ITERS; ++it) {
    const int last = (it == NUM_ITERS - 1);
    row_norm_kernel<<<KCL, 256, 0, stream>>>(mu, mu_n, KCL);
    zero_f32<<<(KCL * DIM + 255) / 256, 256, 0, stream>>>(mu_acc, KCL * DIM);
    zero_f32<<<1, 256, 0, stream>>>(colsum, KCL);
    dist_softmax_kernel<<<N_PAD / 128, 256, 0, stream>>>(
        data_n, mu_n, r_t, colsum, ctemp, last ? out_r : nullptr, last);
    mu_gemm_kernel<<<dim3(KCL / 128, DIM / GDW, N_PAD / CHUNK), 256, 0, stream>>>(
        r_t, data_t, mu_acc);
    finalize_kernel<<<(KCL * DIM) / 256, 256, 0, stream>>>(
        mu_acc, colsum, mu, last ? out_mu : nullptr);
  }
}
